// Attn_71193377898701
// MI455X (gfx1250) — matrix operand order repaired and checked by reading, not compiled, not run on hardware
//
#include <hip/hip_runtime.h>
#include <hip/hip_bf16.h>
#include <math.h>

#define N_EMBD 256
#define N_HEAD 16
#define CTX    2048
#define V_HD   32
#define KV_RANK 32
#define Q_RANK  96
#define ROPE_HD 64
#define NOPE_HD 32
#define HEAD_D  96
#define N_KEEP  512
#define BB 2
#define TT 2048
#define BT (BB*TT)
#define EPSV 1e-6f

typedef __attribute__((ext_vector_type(16))) _Float16 v16h;
typedef __attribute__((ext_vector_type(8)))  _Float16 v8h;
typedef __attribute__((ext_vector_type(8)))  float    v8f;

__device__ __forceinline__ v8f wmma_f16(v16h a, v16h b, v8f c) {
  return __builtin_amdgcn_wmma_f32_16x16x32_f16(false, a, false, b, (short)0, c, false, false);
}

__device__ __forceinline__ v16h cat8(v8h lo, v8h hi) {
  return __builtin_shufflevector(lo, hi, 0,1,2,3,4,5,6,7,8,9,10,11,12,13,14,15);
}

__device__ __forceinline__ float block_reduce_sum256(float v, float* red) {
  int tid = threadIdx.x;
  red[tid] = v; __syncthreads();
  for (int s = 128; s > 0; s >>= 1) { if (tid < s) red[tid] += red[tid + s]; __syncthreads(); }
  float r = red[0]; __syncthreads();
  return r;
}

// ---------------- init ----------------
__global__ void k_init(float* gate_acc) {
  if (threadIdx.x < BB * 3) gate_acc[threadIdx.x] = 0.f;
}

// ---------------- logmap: xh = logmap(shift(x), x, c=1) ----------------
__global__ void k_logmap(const float* __restrict__ x, float* __restrict__ xh) {
  __shared__ float red[256];
  int i = blockIdx.x;            // token
  int b = i / TT, t = i % TT;
  int j = threadIdx.x;
  const float* xu = x + ((size_t)b * TT + t) * N_EMBD;
  float ru = xu[j];
  float rx = (t > 0) ? xu[j - N_EMBD] : 0.f;   // ref = x[b, t-1]
  float xn2 = block_reduce_sum256(rx * rx, red);
  float un2 = block_reduce_sum256(ru * ru, red);
  float ip  = block_reduce_sum256(-rx * ru, red);   // <-x, u>
  const float c = 1.0f;
  float num = (1.f + 2.f * c * ip + c * un2) * (-rx) + (1.f - c * xn2) * ru;
  float den = 1.f + 2.f * c * ip + c * c * xn2 * un2;
  float mob = num / den;
  float an2 = block_reduce_sum256(mob * mob, red);
  float an = sqrtf(an2);
  float cf = (1.f + c * xn2);                     // 2/(sf*sqrt(c)), c=1
  float arg = fminf(sqrtf(an), 0.999f);
  float ath = atanhf(arg);
  float anS = fmaxf(an, 1e-20f);
  xh[((size_t)b * TT + t) * N_EMBD + j] = cf * ath * mob / anS;
}

// ---------------- per-token small projections ----------------
__global__ void k_token_small(const float* __restrict__ x, const float* __restrict__ xh,
                              const float* __restrict__ W_cq, const float* __restrict__ q_norm_w,
                              const float* __restrict__ W_ckv, const float* __restrict__ kv_norm_w,
                              const float* __restrict__ W_kr,
                              const float* __restrict__ W_imp, const float* __restrict__ b_imp,
                              const float* __restrict__ W_gate, const float* __restrict__ b_gate,
                              const float* __restrict__ cos_f, const float* __restrict__ sin_f,
                              float* __restrict__ nq, float* __restrict__ nkv, float* __restrict__ krr,
                              float* __restrict__ scores, float* gate_acc) {
  __shared__ float sx[N_EMBD], sxh[N_EMBD], snq[Q_RANK], snkv[KV_RANK], skr[ROPE_HD], red[256];
  int i = blockIdx.x; int b = i / TT, t = i % TT; int tid = threadIdx.x;
  sx[tid]  = x [((size_t)b * TT + t) * N_EMBD + tid];
  sxh[tid] = xh[((size_t)b * TT + t) * N_EMBD + tid];
  __syncthreads();
  if (tid < Q_RANK) {                 // nq raw = xh @ W_cq
    float acc = 0.f;
    for (int k = 0; k < N_EMBD; k++) acc += sxh[k] * W_cq[k * Q_RANK + tid];
    snq[tid] = acc;
  } else if (tid < Q_RANK + KV_RANK) { // nkv raw = x @ W_ckv
    int j = tid - Q_RANK; float acc = 0.f;
    for (int k = 0; k < N_EMBD; k++) acc += sx[k] * W_ckv[k * KV_RANK + j];
    snkv[j] = acc;
  } else if (tid < Q_RANK + KV_RANK + ROPE_HD) { // kr raw = x @ W_kr
    int j = tid - Q_RANK - KV_RANK; float acc = 0.f;
    for (int k = 0; k < N_EMBD; k++) acc += sx[k] * W_kr[k * ROPE_HD + j];
    skr[j] = acc;
  } else if (tid < 192 + 3) {          // gate logits
    int j = tid - 192; float acc = b_gate[j];
    for (int k = 0; k < N_EMBD; k++) acc += sx[k] * W_gate[k * 3 + j];
    atomicAdd(&gate_acc[b * 3 + j], acc);
  } else if (tid == 195) {             // importance score
    float acc = b_imp[0];
    for (int k = 0; k < N_EMBD; k++) acc += sx[k] * W_imp[k];
    scores[b * TT + t] = acc;
  }
  __syncthreads();
  float vq = (tid < Q_RANK) ? snq[tid] * snq[tid] : 0.f;
  float msq = block_reduce_sum256(vq, red) / (float)Q_RANK;
  float vk = (tid < KV_RANK) ? snkv[tid] * snkv[tid] : 0.f;
  float msk = block_reduce_sum256(vk, red) / (float)KV_RANK;
  if (tid < Q_RANK)
    nq[(size_t)(b * TT + t) * Q_RANK + tid] = snq[tid] * rsqrtf(msq + EPSV) * q_norm_w[tid];
  if (tid < KV_RANK)
    nkv[(size_t)(b * TT + t) * KV_RANK + tid] = snkv[tid] * rsqrtf(msk + EPSV) * kv_norm_w[tid];
  if (tid < 32) {                      // rope on kr
    float cc = cos_f[t * 32 + tid], ss = sin_f[t * 32 + tid];
    float xr = skr[tid], xi = skr[tid + 32];
    krr[(size_t)(b * TT + t) * ROPE_HD + tid]      = xr * cc - xi * ss;
    krr[(size_t)(b * TT + t) * ROPE_HD + tid + 32] = xr * ss + xi * cc;
  }
}

// ---------------- gate softmax ----------------
__global__ void k_gate(const float* gate_acc, float* bw) {
  int b = threadIdx.x;
  if (b < BB) {
    float l0 = gate_acc[b*3+0] / TT, l1 = gate_acc[b*3+1] / TT, l2 = gate_acc[b*3+2] / TT;
    float m = fmaxf(l0, fmaxf(l1, l2));
    float e0 = __expf(l0-m), e1 = __expf(l1-m), e2 = __expf(l2-m);
    float s = e0+e1+e2;
    bw[b*3+0]=e0/s; bw[b*3+1]=e1/s; bw[b*3+2]=e2/s;
  }
}

// ---------------- top-k by rank counting ----------------
__global__ void k_topk(const float* __restrict__ scores, int* __restrict__ idx) {
  __shared__ float ss[TT];
  __shared__ unsigned char keep[TT];
  int b = blockIdx.x, tid = threadIdx.x;
  for (int j = tid; j < TT; j += 256) ss[j] = scores[b * TT + j];
  __syncthreads();
  for (int jt = tid; jt < TT; jt += 256) {
    float sv = ss[jt]; int rank = 0;
    for (int j = 0; j < TT; j++) {
      float o = ss[j];
      rank += (o > sv) || (o == sv && j < jt);
    }
    keep[jt] = (rank < N_KEEP) ? 1 : 0;
  }
  __syncthreads();
  if (tid == 0) {
    int p = 0;
    for (int t2 = 0; t2 < TT; t2++) if (keep[t2] && p < N_KEEP) idx[b * N_KEEP + p++] = t2;
    while (p < N_KEEP) idx[b * N_KEEP + p++] = 0;
  }
}

// ---------------- q / k_cmp / v_cmp generation (V stored dim-major!) ----------------
__global__ void k_qkv(const float* __restrict__ nq, const float* __restrict__ nkv,
                      const float* __restrict__ krr,
                      const float* __restrict__ W_dqn, const float* __restrict__ W_dqr,
                      const float* __restrict__ W_dkn, const float* __restrict__ W_dv,
                      const float* __restrict__ cos_f, const float* __restrict__ sin_f,
                      _Float16* __restrict__ q_h, _Float16* __restrict__ kc_h, _Float16* __restrict__ vc_h) {
  __shared__ float snq[Q_RANK], snkv[KV_RANK], skr[ROPE_HD], sqr[N_HEAD * ROPE_HD];
  int i = blockIdx.x; int b = i / TT, t = i % TT; int tid = threadIdx.x;
  if (tid < Q_RANK)  snq[tid]  = nq [(size_t)i * Q_RANK + tid];
  if (tid < KV_RANK) snkv[tid] = nkv[(size_t)i * KV_RANK + tid];
  if (tid < ROPE_HD) skr[tid]  = krr[(size_t)i * ROPE_HD + tid];
  __syncthreads();
  // qn -> q[...,0:32]
  for (int j = tid; j < N_HEAD * NOPE_HD; j += 256) {
    int h = j >> 5, d = j & 31; float acc = 0.f;
    for (int k = 0; k < Q_RANK; k++) acc += snq[k] * W_dqn[k * (N_HEAD*NOPE_HD) + j];
    q_h[(((size_t)b * N_HEAD + h) * TT + t) * HEAD_D + d] = (_Float16)acc;
  }
  // qr raw
  for (int j = tid; j < N_HEAD * ROPE_HD; j += 256) {
    float acc = 0.f;
    for (int k = 0; k < Q_RANK; k++) acc += snq[k] * W_dqr[k * (N_HEAD*ROPE_HD) + j];
    sqr[j] = acc;
  }
  __syncthreads();
  // qr rope -> q[...,32:96]
  for (int j = tid; j < N_HEAD * ROPE_HD; j += 256) {
    int h = j >> 6, d = j & 63; int dd = d & 31;
    float cc = cos_f[t * 32 + dd], ssn = sin_f[t * 32 + dd];
    float xr = sqr[h * 64 + dd], xi = sqr[h * 64 + 32 + dd];
    float val = (d < 32) ? (xr * cc - xi * ssn) : (xr * ssn + xi * cc);
    q_h[(((size_t)b * N_HEAD + h) * TT + t) * HEAD_D + NOPE_HD + d] = (_Float16)val;
  }
  // kn -> k[...,0:32]
  for (int j = tid; j < N_HEAD * NOPE_HD; j += 256) {
    int h = j >> 5, d = j & 31; float acc = 0.f;
    for (int k = 0; k < KV_RANK; k++) acc += snkv[k] * W_dkn[k * (N_HEAD*NOPE_HD) + j];
    kc_h[(((size_t)b * N_HEAD + h) * TT + t) * HEAD_D + d] = (_Float16)acc;
  }
  // kr broadcast -> k[...,32:96]
  for (int j = tid; j < N_HEAD * ROPE_HD; j += 256) {
    int h = j >> 6, d = j & 63;
    kc_h[(((size_t)b * N_HEAD + h) * TT + t) * HEAD_D + NOPE_HD + d] = (_Float16)skr[d];
  }
  // v: transposed layout [b][h][d][t]
  for (int j = tid; j < N_HEAD * V_HD; j += 256) {
    int h = j >> 5, d = j & 31; float acc = 0.f;
    for (int k = 0; k < KV_RANK; k++) acc += snkv[k] * W_dv[k * (N_HEAD*V_HD) + j];
    vc_h[(((size_t)b * N_HEAD + h) * V_HD + d) * TT + t] = (_Float16)acc;
  }
}

// ---------------- dense [256 -> H*96] + [256 -> H*32] projection from x rows ----------------
__global__ void k_kvdense(const float* __restrict__ x, const int* __restrict__ idx, int useIdx, int rows,
                          const float* __restrict__ Wk, const float* __restrict__ Wv,
                          _Float16* __restrict__ kout, _Float16* __restrict__ vout) {
  __shared__ float sx[N_EMBD];
  int i = blockIdx.x; int b = i / rows, n = i % rows; int tid = threadIdx.x;
  int t = useIdx ? idx[b * rows + n] : n;
  sx[tid] = x[((size_t)b * TT + t) * N_EMBD + tid];
  __syncthreads();
  for (int j = tid; j < N_HEAD * HEAD_D; j += 256) {
    int h = j / HEAD_D, d = j % HEAD_D; float acc = 0.f;
    for (int k = 0; k < N_EMBD; k++) acc += sx[k] * Wk[k * (N_HEAD*HEAD_D) + j];
    kout[(((size_t)b * N_HEAD + h) * rows + n) * HEAD_D + d] = (_Float16)acc;
  }
  // v: transposed layout [b][h][d][rows]
  for (int j = tid; j < N_HEAD * V_HD; j += 256) {
    int h = j >> 5, d = j & 31; float acc = 0.f;
    for (int k = 0; k < N_EMBD; k++) acc += sx[k] * Wv[k * (N_HEAD*V_HD) + j];
    vout[(((size_t)b * N_HEAD + h) * V_HD + d) * rows + n] = (_Float16)acc;
  }
}

// ---------------- flash attention with WMMA (one wave32 per 16-query tile) ----------------
// K layout: [b][h][t][96] (token-major). V layout: [b][h][32][kvT] (dim-major).
__global__ __launch_bounds__(32)
void k_attn(const _Float16* __restrict__ Q, const _Float16* __restrict__ K,
            const _Float16* __restrict__ V, float* __restrict__ O, int kvT, int causal) {
  __shared__ _Float16 ptile[16][40];   // 80B row stride -> 16B-aligned v8h access
  const int NQT = TT / 16;
  int wid = blockIdx.x;
  int qt = wid % NQT; int bh = wid / NQT;
  int b = bh / N_HEAD, h = bh % N_HEAD;
  int qbase = qt * 16;
  int lane = threadIdx.x;
  int row = lane & 15, hi = lane >> 4;

  const _Float16* Qp = Q + (size_t)bh * TT  * HEAD_D;
  const _Float16* Kp = K + (size_t)bh * kvT * HEAD_D;
  const _Float16* Vp = V + (size_t)bh * V_HD * kvT;

  // Q A-fragments (3 chunks of K=32 over HEAD_D=96); 16B-aligned vector loads
  v16h a[3];
  #pragma unroll
  for (int c = 0; c < 3; c++) {
    const _Float16* qr = Qp + (size_t)(qbase + row) * HEAD_D + c * 32;
    a[c] = cat8(*(const v8h*)(qr + 16 * hi), *(const v8h*)(qr + 16 * hi + 8));
  }

  float m_i[8], l_i[8];
  #pragma unroll
  for (int r = 0; r < 8; r++) { m_i[r] = -INFINITY; l_i[r] = 0.f; }
  v8f o0 = {0.f,0.f,0.f,0.f,0.f,0.f,0.f,0.f};
  v8f o1 = {0.f,0.f,0.f,0.f,0.f,0.f,0.f,0.f};

  const float scale = 0.1020620726159658f; // 1/sqrt(96)
  int kend = causal ? (qbase + 16 < kvT ? qbase + 16 : kvT) : kvT;

  for (int kb = 0; kb < kend; kb += 32) {
    int key0 = kb + row, key1 = kb + 16 + row;
    int ck0 = key0 < kvT ? key0 : kvT - 1;
    int ck1 = key1 < kvT ? key1 : kvT - 1;
    const _Float16* krow0 = Kp + (size_t)ck0 * HEAD_D + 16 * hi;
    const _Float16* krow1 = Kp + (size_t)ck1 * HEAD_D + 16 * hi;
    // prefetch next K tile (streams through cache hierarchy)
    if (kb + 32 < kend) __builtin_prefetch(krow0 + 32 * HEAD_D, 0, 0);

    // ---- issue ALL tile loads up front (one big clause; latency overlapped below) ----
    v16h bk0[3], bk1[3];
    #pragma unroll
    for (int c = 0; c < 3; c++) {
      bk0[c] = cat8(*(const v8h*)(krow0 + c * 32), *(const v8h*)(krow0 + c * 32 + 8));
      bk1[c] = cat8(*(const v8h*)(krow1 + c * 32), *(const v8h*)(krow1 + c * 32 + 8));
    }
    const _Float16* vp0 = Vp + (size_t)row        * kvT + kb + 16 * hi;
    const _Float16* vp1 = Vp + (size_t)(16 + row) * kvT + kb + 16 * hi;
    v16h vb0 = cat8(*(const v8h*)vp0, *(const v8h*)(vp0 + 8));
    v16h vb1 = cat8(*(const v8h*)vp1, *(const v8h*)(vp1 + 8));

    // ---- S = Q K^T for 32-key tile (two 16-key subtiles) ----
    v8f s0 = {0.f,0.f,0.f,0.f,0.f,0.f,0.f,0.f};
    v8f s1 = {0.f,0.f,0.f,0.f,0.f,0.f,0.f,0.f};
    #pragma unroll
    for (int c = 0; c < 3; c++) {
      s0 = wmma_f16(a[c], bk0[c], s0);
      s1 = wmma_f16(a[c], bk1[c], s1);
    }

    // ---- online softmax (per output row m = r + 8*hi; cols spread over 16 lanes) ----
    #pragma unroll
    for (int r = 0; r < 8; r++) {
      int m = r + 8 * hi;
      int qidx = qbase + m;
      int k0 = kb + row, k1 = k0 + 16;
      float v0 = s0[r] * scale, v1 = s1[r] * scale;
      bool ok0 = (k0 < kvT) && (!causal || k0 <= qidx);
      bool ok1 = (k1 < kvT) && (!causal || k1 <= qidx);
      v0 = ok0 ? v0 : -1e30f;
      v1 = ok1 ? v1 : -1e30f;
      float mx = fmaxf(v0, v1);
      for (int off = 1; off < 16; off <<= 1) mx = fmaxf(mx, __shfl_xor(mx, off, 32));
      float mn = fmaxf(m_i[r], mx);
      float p0 = __expf(v0 - mn), p1 = __expf(v1 - mn);
      float ps = p0 + p1;
      for (int off = 1; off < 16; off <<= 1) ps += __shfl_xor(ps, off, 32);
      float corr = __expf(m_i[r] - mn);
      l_i[r] = l_i[r] * corr + ps;
      m_i[r] = mn;
      o0[r] *= corr; o1[r] *= corr;
      ptile[m][row]      = (_Float16)p0;
      ptile[m][16 + row] = (_Float16)p1;
    }
    __syncthreads();
    // P A-fragment from LDS (two aligned b128 reads per lane), then O += P V
    const _Float16* pr = &ptile[row][0];
    v16h pa = cat8(*(const v8h*)(pr + 16 * hi), *(const v8h*)(pr + 16 * hi + 8));
    o0 = wmma_f16(pa, vb0, o0);
    o1 = wmma_f16(pa, vb1, o1);
    __syncthreads();
  }
  // epilogue: O layout [b][t][h][vd]
  #pragma unroll
  for (int r = 0; r < 8; r++) {
    int m = r + 8 * hi;
    float inv = 1.0f / l_i[r];
    size_t base = (((size_t)b * TT + (qbase + m)) * N_HEAD + h) * V_HD;
    O[base + row]      = o0[r] * inv;
    O[base + 16 + row] = o1[r] * inv;
  }
}

// ---------------- combine branches + output projection ----------------
__global__ void k_proj(const float* __restrict__ o_cmp, const float* __restrict__ o_sel,
                       const float* __restrict__ o_win, const float* __restrict__ bw,
                       const float* __restrict__ W_proj, float* __restrict__ out) {
  __shared__ float comb[N_HEAD * V_HD];
  int i = blockIdx.x; int b = i / TT; int tid = threadIdx.x;
  float w0 = bw[b*3+0], w1 = bw[b*3+1], w2 = bw[b*3+2];
  size_t base = (size_t)i * (N_HEAD * V_HD);
  for (int j = tid; j < N_HEAD * V_HD; j += 256)
    comb[j] = w0 * o_cmp[base + j] + w1 * o_sel[base + j] + w2 * o_win[base + j];
  __syncthreads();
  float acc = 0.f;
  for (int k = 0; k < N_HEAD * V_HD; k++) acc += comb[k] * W_proj[k * N_EMBD + tid];
  out[(size_t)i * N_EMBD + tid] = acc;
}

extern "C" void kernel_launch(void* const* d_in, const int* in_sizes, int n_in,
                              void* d_out, int out_size, void* d_ws, size_t ws_size,
                              hipStream_t stream) {
  const float* x        = (const float*)d_in[0];
  const float* W_cq     = (const float*)d_in[1];
  const float* q_norm_w = (const float*)d_in[2];
  const float* W_dqn    = (const float*)d_in[3];
  const float* W_dqr    = (const float*)d_in[4];
  const float* W_ckv    = (const float*)d_in[5];
  const float* kv_norm_w= (const float*)d_in[6];
  const float* W_dkn    = (const float*)d_in[7];
  const float* W_dv     = (const float*)d_in[8];
  const float* W_kr     = (const float*)d_in[9];
  const float* W_imp    = (const float*)d_in[10];
  const float* b_imp    = (const float*)d_in[11];
  const float* W_selk   = (const float*)d_in[12];
  const float* W_selv   = (const float*)d_in[13];
  const float* W_wink   = (const float*)d_in[14];
  const float* W_winv   = (const float*)d_in[15];
  const float* W_gate   = (const float*)d_in[16];
  const float* b_gate   = (const float*)d_in[17];
  const float* W_proj   = (const float*)d_in[18];
  const float* cos_f    = (const float*)d_in[19];
  const float* sin_f    = (const float*)d_in[20];
  float* out = (float*)d_out;

  char* base = (char*)d_ws; size_t off = 0;
  auto carve = [&](size_t bytes) -> void* {
    void* p = base + off; off = (off + bytes + 255) & ~(size_t)255; return p;
  };
  float*    xh     = (float*)   carve((size_t)BT * N_EMBD * 4);
  float*    nq     = (float*)   carve((size_t)BT * Q_RANK * 4);
  float*    nkv    = (float*)   carve((size_t)BT * KV_RANK * 4);
  float*    krr    = (float*)   carve((size_t)BT * ROPE_HD * 4);
  float*    scores = (float*)   carve((size_t)BT * 4);
  float*    gacc   = (float*)   carve(BB * 3 * 4);
  float*    bw     = (float*)   carve(BB * 3 * 4);
  int*      idx    = (int*)     carve((size_t)BB * N_KEEP * 4);
  _Float16* q_h    = (_Float16*)carve((size_t)BB * N_HEAD * TT * HEAD_D * 2);
  _Float16* kc_h   = (_Float16*)carve((size_t)BB * N_HEAD * TT * HEAD_D * 2);
  _Float16* vc_h   = (_Float16*)carve((size_t)BB * N_HEAD * TT * V_HD * 2);
  _Float16* wk_h   = (_Float16*)carve((size_t)BB * N_HEAD * TT * HEAD_D * 2);
  _Float16* wv_h   = (_Float16*)carve((size_t)BB * N_HEAD * TT * V_HD * 2);
  _Float16* sk_h   = (_Float16*)carve((size_t)BB * N_HEAD * N_KEEP * HEAD_D * 2);
  _Float16* sv_h   = (_Float16*)carve((size_t)BB * N_HEAD * N_KEEP * V_HD * 2);
  float*    o_cmp  = (float*)   carve((size_t)BT * N_HEAD * V_HD * 4);
  float*    o_sel  = (float*)   carve((size_t)BT * N_HEAD * V_HD * 4);
  float*    o_win  = (float*)   carve((size_t)BT * N_HEAD * V_HD * 4);

  k_init<<<1, 32, 0, stream>>>(gacc);
  k_logmap<<<BT, 256, 0, stream>>>(x, xh);
  k_token_small<<<BT, 256, 0, stream>>>(x, xh, W_cq, q_norm_w, W_ckv, kv_norm_w, W_kr,
                                        W_imp, b_imp, W_gate, b_gate, cos_f, sin_f,
                                        nq, nkv, krr, scores, gacc);
  k_gate<<<1, 32, 0, stream>>>(gacc, bw);
  k_topk<<<BB, 256, 0, stream>>>(scores, idx);
  k_qkv<<<BT, 256, 0, stream>>>(nq, nkv, krr, W_dqn, W_dqr, W_dkn, W_dv, cos_f, sin_f,
                                q_h, kc_h, vc_h);
  k_kvdense<<<BT, 256, 0, stream>>>(x, idx, 0, TT, W_wink, W_winv, wk_h, wv_h);
  k_kvdense<<<BB * N_KEEP, 256, 0, stream>>>(x, idx, 1, N_KEEP, W_selk, W_selv, sk_h, sv_h);

  int nblk = BB * N_HEAD * (TT / 16);
  k_attn<<<nblk, 32, 0, stream>>>(q_h, kc_h, vc_h, o_cmp, TT, 1);
  k_attn<<<nblk, 32, 0, stream>>>(q_h, sk_h, sv_h, o_sel, N_KEEP, 0);
  k_attn<<<nblk, 32, 0, stream>>>(q_h, wk_h, wv_h, o_win, TT, 1);

  k_proj<<<BT, 256, 0, stream>>>(o_cmp, o_sel, o_win, bw, W_proj, out);
  (void)in_sizes; (void)n_in; (void)out_size; (void)ws_size;
}
